// Self_Attn_20486994002729
// MI455X (gfx1250) — matrix operand order repaired and checked by reading, not compiled, not run on hardware
//
#include <hip/hip_runtime.h>

// ---------------------------------------------------------------------------
// Self-attention (B=8, C=256, N=2048, DQK=32) for gfx1250 (MI455X, wave32).
// Prep:   x -> x^T bf16, weights -> bf16 (one-time converts).
// Phase1: q,k,v projections, pure bf16 WMMA with contiguous b128 loads.
// Phase2: flash attention; K/V strips staged via GLOBAL_LOAD_ASYNC_TO_LDS
//         (double-buffered, ASYNCcnt), online softmax, fused gamma*out + x.
// ---------------------------------------------------------------------------

typedef __bf16 bf16;
typedef bf16  v8bf  __attribute__((ext_vector_type(8)));
typedef bf16  v16bf __attribute__((ext_vector_type(16)));
typedef float v8f   __attribute__((ext_vector_type(8)));

constexpr int kB = 8;
constexpr int kC = 256;
constexpr int kN = 2048;
constexpr int kD = 32;                         // DQK
constexpr int kWaves = 4;                      // waves per block
constexpr int kTilesTotal = kB * (kN / 16);    // 1024 tiles of 16 rows
constexpr int kStride = 40;                    // padded LDS row stride (bf16)

__device__ __forceinline__ v16bf cat16(v8bf lo, v8bf hi) {
    return __builtin_shufflevector(lo, hi, 0,1,2,3,4,5,6,7,8,9,10,11,12,13,14,15);
}
__device__ __forceinline__ v16bf ldg16(const bf16* p) {           // 2 x b128
    return cat16(*(const v8bf*)p, *(const v8bf*)(p + 8));
}

// CDNA5 async DMA: 16B global -> LDS, tracked by ASYNCcnt (no VGPR transit).
// LDS byte address = low 32 bits of the generic pointer (flat LDS aperture).
__device__ __forceinline__ void async_ld128(const bf16* lds_p, const bf16* g_p) {
    const unsigned lds_addr = (unsigned)(size_t)lds_p;
    asm volatile("global_load_async_to_lds_b128 %0, %1, off"
                 :: "v"(lds_addr), "v"(g_p) : "memory");
}
__device__ __forceinline__ void wait_async0() {
    asm volatile("s_wait_asynccnt 0x0" ::: "memory");
}

// ---------------------------------------------------------------------------
// Prep A: x (B,C,N) f32 -> xb (B,N,C) bf16, 32x32 tiles through LDS.
// ---------------------------------------------------------------------------
__global__ __launch_bounds__(128) void xpose_kernel(
    const float* __restrict__ x, bf16* __restrict__ xb)
{
    __shared__ float tile[32][33];
    const int tilesPerB = (kN / 32) * (kC / 32);       // 64 * 8
    const int b   = blockIdx.x / tilesPerB;
    const int rem = blockIdx.x % tilesPerB;
    const int n0  = (rem / (kC / 32)) * 32;
    const int c0  = (rem % (kC / 32)) * 32;
    const int t   = threadIdx.x;

    {   // coalesced read: row c = c0+i, 8 consecutive n per thread
        const int i  = t >> 2;
        const int ch = (t & 3) * 8;
        const float* p = x + ((size_t)b * kC + c0 + i) * kN + n0 + ch;
        const float4 a0 = *(const float4*)p;
        const float4 a1 = *(const float4*)(p + 4);
        tile[i][ch + 0] = a0.x; tile[i][ch + 1] = a0.y;
        tile[i][ch + 2] = a0.z; tile[i][ch + 3] = a0.w;
        tile[i][ch + 4] = a1.x; tile[i][ch + 5] = a1.y;
        tile[i][ch + 6] = a1.z; tile[i][ch + 7] = a1.w;
    }
    __syncthreads();
    {   // coalesced write: row n = n0+j, 8 consecutive c per thread (bf16)
        const int j  = t >> 2;
        const int ch = (t & 3) * 8;
        v8bf o;
        #pragma unroll
        for (int e = 0; e < 8; ++e) o[e] = (bf16)tile[ch + e][j];
        *(v8bf*)(xb + ((size_t)b * kN + n0 + j) * kC + c0 + ch) = o;
    }
}

// ---------------------------------------------------------------------------
// Prep B: weights f32 -> bf16 (row-major layout preserved).
// ---------------------------------------------------------------------------
__global__ __launch_bounds__(128) void wconv_kernel(
    const float* __restrict__ wq, const float* __restrict__ wk,
    const float* __restrict__ wv,
    bf16* __restrict__ wbq, bf16* __restrict__ wbk, bf16* __restrict__ wbv)
{
    const int i = blockIdx.x * blockDim.x + threadIdx.x;   // 8192 threads
    if (i < kD * kC) { wbq[i] = (bf16)wq[i]; wbk[i] = (bf16)wk[i]; }
    for (int j = i; j < kC * kC; j += gridDim.x * blockDim.x)
        wbv[j] = (bf16)wv[j];
}

// ---------------------------------------------------------------------------
// Phase 1: q = Wq x + bq (B,N,32), k likewise, v = Wv x + bv (B,C,N), bf16.
// One wave per 16-row n-tile; all fragment loads are contiguous bf16 b128.
// ---------------------------------------------------------------------------
__global__ __launch_bounds__(kWaves * 32) void proj_kernel(
    const bf16* __restrict__ xb,
    const bf16* __restrict__ wbq, const float* __restrict__ bq,
    const bf16* __restrict__ wbk, const float* __restrict__ bk,
    const bf16* __restrict__ wbv, const float* __restrict__ bv,
    bf16* __restrict__ qb, bf16* __restrict__ kb, bf16* __restrict__ vb)
{
    const int wave = threadIdx.x >> 5;
    const int lane = threadIdx.x & 31;
    const int col  = lane & 15;
    const int half = lane >> 4;
    const int tile = blockIdx.x * kWaves + wave;
    const int b    = tile / (kN / 16);
    const int n0   = (tile % (kN / 16)) * 16;

    const v8f vzero = {0,0,0,0,0,0,0,0};
    v8f qacc[2], kacc[2], vacc[16];
    #pragma unroll
    for (int i = 0; i < 2; ++i) { qacc[i] = vzero; kacc[i] = vzero; }
    #pragma unroll
    for (int i = 0; i < 16; ++i) vacc[i] = vzero;

    const bf16* arow = xb + ((size_t)b * kN + n0 + col) * kC;   // A row n0+col

    for (int c0 = 0; c0 < kC; c0 += 32) {
        // A-frag: lanes<16 hold K {0..15}, lanes>=16 hold {16..31}
        const bf16* ap = arow + c0 + half * 16;
        const v16bf a = cat16(*(const v8bf*)ap, *(const v8bf*)(ap + 8));

        #pragma unroll
        for (int dt = 0; dt < 2; ++dt) {
            const v16bf bq16 = ldg16(wbq + (size_t)(dt*16 + col) * kC + c0 + half*16);
            const v16bf bk16 = ldg16(wbk + (size_t)(dt*16 + col) * kC + c0 + half*16);
            qacc[dt] = __builtin_amdgcn_wmma_f32_16x16x32_bf16(
                false, a, false, bq16, (short)0, qacc[dt], false, false);
            kacc[dt] = __builtin_amdgcn_wmma_f32_16x16x32_bf16(
                false, a, false, bk16, (short)0, kacc[dt], false, false);
        }
        #pragma unroll
        for (int ct = 0; ct < 16; ++ct) {
            const v16bf bv16 = ldg16(wbv + (size_t)(ct*16 + col) * kC + c0 + half*16);
            vacc[ct] = __builtin_amdgcn_wmma_f32_16x16x32_bf16(
                false, a, false, bv16, (short)0, vacc[ct], false, false);
        }
    }

    #pragma unroll
    for (int dt = 0; dt < 2; ++dt) {
        const int d = dt * 16 + col;
        const float biasq = bq[d];
        const float biask = bk[d];
        #pragma unroll
        for (int r = 0; r < 8; ++r) {
            const int nn = n0 + r + 8 * half;
            qb[((size_t)b * kN + nn) * kD + d] = (bf16)(qacc[dt][r] + biasq);
            kb[((size_t)b * kN + nn) * kD + d] = (bf16)(kacc[dt][r] + biask);
        }
    }
    #pragma unroll
    for (int ct = 0; ct < 16; ++ct) {
        const int c = ct * 16 + col;
        const float bias = bv[c];
        #pragma unroll
        for (int r = 0; r < 8; ++r) {
            const int nn = n0 + r + 8 * half;
            vb[((size_t)b * kC + c) * kN + nn] = (bf16)(vacc[ct][r] + bias);
        }
    }
}

// ---------------------------------------------------------------------------
// Phase 2: flash attention. Block = 4 waves = 4 query tiles, same batch.
// K/V strips (32 keys) DMA'd into double-buffered LDS via async-to-LDS;
// next strip's DMA is issued before the current strip's 18 WMMAs.
// ---------------------------------------------------------------------------
__global__ __launch_bounds__(kWaves * 32) void attn_kernel(
    const bf16* __restrict__ qb, const bf16* __restrict__ kb,
    const bf16* __restrict__ vb, const float* __restrict__ x,
    const float* __restrict__ gamma_p, float* __restrict__ out)
{
    __shared__ __align__(16) bf16 vlds[2][kC * kStride];   // 2 x 20 KiB
    __shared__ __align__(16) bf16 klds[2][32 * kStride];   // 2 x 2.5 KiB
    __shared__ __align__(16) bf16 pbuf[kWaves][16 * 32];   // 4 KiB

    const int t    = threadIdx.x;                 // 0..127
    const int wave = t >> 5;
    const int lane = t & 31;
    const int col  = lane & 15;
    const int half = lane >> 4;
    const int b    = blockIdx.x >> 5;                         // 32 blocks/batch
    const int i0   = ((blockIdx.x & 31) * kWaves + wave) * 16;

    // ---- Q A-frag, held in registers for the whole loop ----
    const bf16* qrow = qb + ((size_t)b * kN + i0 + col) * kD + half * 16;
    const v16bf qfrag = cat16(*(const v8bf*)qrow, *(const v8bf*)(qrow + 8));

    const v8f vzero = {0,0,0,0,0,0,0,0};
    v8f acc[16];
    #pragma unroll
    for (int i = 0; i < 16; ++i) acc[i] = vzero;
    float m[8], l[8];
    #pragma unroll
    for (int r = 0; r < 8; ++r) { m[r] = -3.0e38f; l[r] = 0.0f; }

    bf16* pw = &pbuf[wave][0];
    const bf16* prow = pw + col * 32 + half * 16;

    const size_t vbb = (size_t)b * kC * kN;
    const size_t kbb = (size_t)b * kN * (size_t)kD;

    // cooperative async strip staging (V: 1024 16B chunks, K: 128 16B chunks)
    auto stage_strips = [&](int j0, int buf) {
        #pragma unroll
        for (int i = 0; i < 8; ++i) {
            const int id = t + 128 * i;
            const int c  = id >> 2;
            const int jc = id & 3;
            async_ld128(&vlds[buf][c * kStride + jc * 8],
                        vb + vbb + (size_t)c * kN + j0 + jc * 8);
        }
        async_ld128(&klds[buf][(t >> 2) * kStride + (t & 3) * 8],
                    kb + kbb + (size_t)(t >> 2) * kD + (size_t)j0 * kD + (t & 3) * 8);
    };

    stage_strips(0, 0);
    wait_async0();
    __syncthreads();

    for (int j0 = 0; j0 < kN; j0 += 32) {
        const int buf = (j0 >> 5) & 1;
        const bool more = (j0 + 32) < kN;
        if (more) stage_strips(j0 + 32, buf ^ 1);   // DMA overlaps WMMA below
        if (j0 + 64 < kN)                           // hint strip after next
            __builtin_prefetch(vb + vbb + (size_t)(t >> 2) * kN + j0 + 64, 0, 1);

        // ---- S = Q K^T : two 16x16 tiles from LDS K strip ----
        v8f s[2];
        #pragma unroll
        for (int jt = 0; jt < 2; ++jt) {
            const bf16* kr = &klds[buf][(jt * 16 + col) * kStride + half * 16];
            s[jt] = __builtin_amdgcn_wmma_f32_16x16x32_bf16(
                false, qfrag, false, ldg16(kr), (short)0, vzero, false, false);
        }

        // ---- online softmax (lane holds 8 row-stats, dup across 16 lanes) --
        float tmax[8];
        #pragma unroll
        for (int r = 0; r < 8; ++r) tmax[r] = fmaxf(s[0][r], s[1][r]);
        #pragma unroll
        for (int off = 1; off < 16; off <<= 1)
            #pragma unroll
            for (int r = 0; r < 8; ++r)
                tmax[r] = fmaxf(tmax[r], __shfl_xor(tmax[r], off, 32));

        v8f corrv;
        #pragma unroll
        for (int r = 0; r < 8; ++r) {
            const float mn = fmaxf(m[r], tmax[r]);
            corrv[r] = __expf(m[r] - mn);
            m[r] = mn;
        }
        float psum[8];
        #pragma unroll
        for (int r = 0; r < 8; ++r) {
            s[0][r] = __expf(s[0][r] - m[r]);
            s[1][r] = __expf(s[1][r] - m[r]);
            psum[r] = s[0][r] + s[1][r];
        }
        #pragma unroll
        for (int off = 1; off < 16; off <<= 1)
            #pragma unroll
            for (int r = 0; r < 8; ++r)
                psum[r] += __shfl_xor(psum[r], off, 32);
        #pragma unroll
        for (int r = 0; r < 8; ++r) l[r] = l[r] * corrv[r] + psum[r];

        // vector rescale of accumulators (packed f32 muls)
        #pragma unroll
        for (int ct = 0; ct < 16; ++ct) acc[ct] *= corrv;

        // ---- transpose P (f32 D-layout) -> bf16 A-layout via per-wave LDS --
        #pragma unroll
        for (int r = 0; r < 8; ++r) {
            const int row = r + 8 * half;
            pw[row * 32 + col]      = (bf16)s[0][r];
            pw[row * 32 + 16 + col] = (bf16)s[1][r];
        }
        asm volatile("s_wait_dscnt 0x0" ::: "memory");   // same-wave LDS RAW
        const v16bf pf = cat16(*(const v8bf*)prow, *(const v8bf*)(prow + 8));

        // ---- acc += P.V over 16 channel tiles from LDS V strip ----
        #pragma unroll
        for (int ct = 0; ct < 16; ++ct) {
            const bf16* vr = &vlds[buf][(ct * 16 + col) * kStride + half * 16];
            acc[ct] = __builtin_amdgcn_wmma_f32_16x16x32_bf16(
                false, pf, false, ldg16(vr), (short)0, acc[ct], false, false);
        }

        if (more) wait_async0();     // back buffer fully in LDS
        __syncthreads();
    }

    // ---- epilogue: normalize, gamma * out + x, transposed store ----
    const float g = gamma_p[0];
    float inv[8];
    #pragma unroll
    for (int r = 0; r < 8; ++r) inv[r] = 1.0f / l[r];

    #pragma unroll
    for (int ct = 0; ct < 16; ++ct) {
        const int c = ct * 16 + col;
        const size_t base = ((size_t)b * kC + c) * kN + i0 + 8 * half;
        const float4 x0 = *(const float4*)(x + base);
        const float4 x1 = *(const float4*)(x + base + 4);
        float4 r0, r1;
        r0.x = g * acc[ct][0] * inv[0] + x0.x;
        r0.y = g * acc[ct][1] * inv[1] + x0.y;
        r0.z = g * acc[ct][2] * inv[2] + x0.z;
        r0.w = g * acc[ct][3] * inv[3] + x0.w;
        r1.x = g * acc[ct][4] * inv[4] + x1.x;
        r1.y = g * acc[ct][5] * inv[5] + x1.y;
        r1.z = g * acc[ct][6] * inv[6] + x1.z;
        r1.w = g * acc[ct][7] * inv[7] + x1.w;
        *(float4*)(out + base)     = r0;
        *(float4*)(out + base + 4) = r1;
    }
}

// ---------------------------------------------------------------------------
extern "C" void kernel_launch(void* const* d_in, const int* in_sizes, int n_in,
                              void* d_out, int out_size, void* d_ws, size_t ws_size,
                              hipStream_t stream) {
    const float* x     = (const float*)d_in[0];
    const float* wq    = (const float*)d_in[1];
    const float* bq    = (const float*)d_in[2];
    const float* wk    = (const float*)d_in[3];
    const float* bk    = (const float*)d_in[4];
    const float* wv    = (const float*)d_in[5];
    const float* bv    = (const float*)d_in[6];
    const float* gamma = (const float*)d_in[7];
    float* out = (float*)d_out;

    char* wsp = (char*)d_ws;
    const size_t xbBytes = (size_t)kB * kN * kC * sizeof(bf16);   // 8 MiB
    const size_t qkBytes = (size_t)kB * kN * kD * sizeof(bf16);   // 1 MiB
    const size_t vBytes  = (size_t)kB * kC * kN * sizeof(bf16);   // 8 MiB
    const size_t wqkB    = (size_t)kD * kC * sizeof(bf16);        // 16 KiB
    bf16* xb  = (bf16*)(wsp);
    bf16* qb  = (bf16*)(wsp + xbBytes);
    bf16* kbp = (bf16*)(wsp + xbBytes + qkBytes);
    bf16* vb  = (bf16*)(wsp + xbBytes + 2 * qkBytes);
    bf16* wbq = (bf16*)(wsp + xbBytes + 2 * qkBytes + vBytes);
    bf16* wbk = (bf16*)(wsp + xbBytes + 2 * qkBytes + vBytes + wqkB);
    bf16* wbv = (bf16*)(wsp + xbBytes + 2 * qkBytes + vBytes + 2 * wqkB);

    xpose_kernel<<<kB * (kN / 32) * (kC / 32), 128, 0, stream>>>(x, xb);
    wconv_kernel<<<64, 128, 0, stream>>>(wq, wk, wv, wbq, wbk, wbv);

    const int blocks = kTilesTotal / kWaves;   // 256
    proj_kernel<<<blocks, kWaves * 32, 0, stream>>>(xb, wbq, bq, wbk, bk,
                                                    wbv, bv, qb, kbp, vb);
    attn_kernel<<<blocks, kWaves * 32, 0, stream>>>(qb, kbp, vb, x, gamma, out);
}
